// VectorQuantizer_17317308138063
// MI455X (gfx1250) — matrix operand order repaired and checked by reading, not compiled, not run on hardware
//
#include <hip/hip_runtime.h>
#include <hip/hip_bf16.h>

// ---------------------------------------------------------------------------
// VectorQuantizer forward for MI455X (gfx1250, wave32, WMMA)
//   inputs:   [65536, 256] f32
//   codebook: [ 8192, 256] f32
//   out:      quantized_st [65536,256] f32  ++  loss scalar f32
// ---------------------------------------------------------------------------

typedef _Float16 half16 __attribute__((ext_vector_type(16)));
typedef _Float16 half8  __attribute__((ext_vector_type(8)));
typedef float    f32x8  __attribute__((ext_vector_type(8)));

#define VQ_D 256
#define LDS_STRIDE 264   // 8 halves pad/row -> conflict-free column reads

#if __has_builtin(__builtin_amdgcn_global_load_async_to_lds_b128) && \
    __has_builtin(__builtin_amdgcn_s_wait_asynccnt)
#define VQ_ASYNC 1
// Builtin signature (probed): (int4 AS1*, int4 AS3*, imm offset, imm cpol)
typedef int v4i_ls __attribute__((vector_size(16)));
typedef __attribute__((address_space(1))) v4i_ls* vq_gptr_t;
typedef __attribute__((address_space(3))) v4i_ls* vq_sptr_t;
#define VQ_ASYNC_B128(gp, sp) \
  __builtin_amdgcn_global_load_async_to_lds_b128((vq_gptr_t)(gp), (vq_sptr_t)(sp), 0, 0)
#else
#define VQ_ASYNC 0
#endif

#if __has_builtin(__builtin_amdgcn_sched_group_barrier)
// Pin tile body to: 16 ds_load (one clause, one wait) then 8 back-to-back WMMA.
#define VQ_SCHED_PIPELINE() do {                                   \
    __builtin_amdgcn_sched_group_barrier(0x100, 16, 0); /* DS rd */ \
    __builtin_amdgcn_sched_group_barrier(0x008,  8, 0); /* WMMA  */ \
  } while (0)
#else
#define VQ_SCHED_PIPELINE()
#endif

// ---------------------------------------------------------------------------
// Kernel 1: row-wise L2 normalize f32 -> f16.  One wave per row (8 rows/block).
// ---------------------------------------------------------------------------
__global__ __launch_bounds__(256) void vq_normalize(
    const float* __restrict__ src, _Float16* __restrict__ dst, int rows) {
  const int lane = threadIdx.x & 31;
  const int row  = (int)((blockIdx.x * blockDim.x + threadIdx.x) >> 5);
  if (row >= rows) return;

  const float* s = src + (size_t)row * VQ_D + lane * 8;
  float4 v0 = *(const float4*)(s);
  float4 v1 = *(const float4*)(s + 4);

  float ss = v0.x * v0.x + v0.y * v0.y + v0.z * v0.z + v0.w * v0.w +
             v1.x * v1.x + v1.y * v1.y + v1.z * v1.z + v1.w * v1.w;
#pragma unroll
  for (int m = 1; m < 32; m <<= 1) ss += __shfl_xor(ss, m, 32);

  const float scale = 1.0f / fmaxf(sqrtf(ss), 1e-12f);

  half8 h;
  h[0] = (_Float16)(v0.x * scale); h[1] = (_Float16)(v0.y * scale);
  h[2] = (_Float16)(v0.z * scale); h[3] = (_Float16)(v0.w * scale);
  h[4] = (_Float16)(v1.x * scale); h[5] = (_Float16)(v1.y * scale);
  h[6] = (_Float16)(v1.z * scale); h[7] = (_Float16)(v1.w * scale);
  *(half8*)(dst + (size_t)row * VQ_D + lane * 8) = h;
}

// ---------------------------------------------------------------------------
// Kernel 2: WMMA cosine-similarity sweep + argmax.
//   Block = 128 threads = 4 waves; each wave owns a 16-row M tile with the full
//   16x256 f16 A tile in registers.  Codebook tiles (16 rows) are staged in
//   double-buffered LDS via async global->LDS loads; the next-tile fetch is
//   issued before the WMMA chain so global latency hides behind matrix work.
//   Two interleaved accumulator chains keep the matrix pipe dependency-free.
// ---------------------------------------------------------------------------
__global__ __launch_bounds__(128) void vq_argmax(
    const _Float16* __restrict__ inp_n, const _Float16* __restrict__ cb_n,
    int* __restrict__ indices, int K) {
  __shared__ __align__(16) _Float16 lds_cb[2 * 16 * LDS_STRIDE];

  const int tid    = threadIdx.x;
  const int lane   = tid & 31;
  const int wave   = tid >> 5;       // 0..3
  const int n16    = lane & 15;      // row (for A) / column (for B,C/D)
  const int hi     = lane >> 4;      // half-wave select
  const int m_base = ((int)blockIdx.x * 4 + wave) * 16;

  // --- Preload A tile: lane holds row (m_base + n16), 8 chunks of K=32 ------
  // Layout (ISA 7.12.2, 16-bit A 16x32): halves 0-7 = K 32c+8*hi,
  //                                      halves 8-15 = K 32c+16+8*hi
  half16 a[8];
  {
    const _Float16* arow = inp_n + (size_t)(m_base + n16) * VQ_D;
#pragma unroll
    for (int c = 0; c < 8; ++c) {
      half8 lo = *(const half8*)(arow + 32 * c + 8 * hi);
      half8 hh = *(const half8*)(arow + 32 * c + 16 + 8 * hi);
      a[c] = __builtin_shufflevector(lo, hh, 0, 1, 2, 3, 4, 5, 6, 7,
                                     8, 9, 10, 11, 12, 13, 14, 15);
    }
  }

  float best[8];
  int   bidx[8];
#pragma unroll
  for (int v = 0; v < 8; ++v) { best[v] = -2.0f; bidx[v] = 0; }

  // Per-tile compute: 16 DS reads -> 8 WMMAs (2 chains) -> argmax update.
  auto tile_compute = [&](const _Float16* bufc, int k0) {
    half16 b[8];
#pragma unroll
    for (int c = 0; c < 8; ++c) {
      half8 blo = *(const half8*)(bufc + n16 * LDS_STRIDE + 32 * c + 8 * hi);
      half8 bhh = *(const half8*)(bufc + n16 * LDS_STRIDE + 32 * c + 16 + 8 * hi);
      b[c] = __builtin_shufflevector(blo, bhh, 0, 1, 2, 3, 4, 5, 6, 7,
                                     8, 9, 10, 11, 12, 13, 14, 15);
    }

    f32x8 acc0 = {}, acc1 = {};
#pragma unroll
    for (int c = 0; c < 8; c += 2) {
      acc0 = __builtin_amdgcn_wmma_f32_16x16x32_f16(
          false, a[c], false, b[c], (short)0, acc0, false, false);
      acc1 = __builtin_amdgcn_wmma_f32_16x16x32_f16(
          false, a[c + 1], false, b[c + 1], (short)0, acc1, false, false);
    }
    VQ_SCHED_PIPELINE();

    const int cand = k0 + n16;
#pragma unroll
    for (int v = 0; v < 8; ++v) {
      const float sv = acc0[v] + acc1[v];
      if (sv > best[v]) { best[v] = sv; bidx[v] = cand; }
    }
  };

  // Cooperative-fetch geometry: 16 rows x 256 halves (8 KB) = 64 B / thread.
  const int r   = tid >> 3;         // codebook row within tile, 0..15
  const int off = (tid & 7) * 32;   // half offset within row, 0..224
  const _Float16* gsrc = cb_n + (size_t)r * VQ_D + off;
  _Float16* sbuf[2] = { lds_cb + r * LDS_STRIDE + off,
                        lds_cb + (16 + r) * LDS_STRIDE + off };

  const int ntiles = K >> 4;        // 512

  // --- Prologue: fetch tile 0 into buffer 0 --------------------------------
#if VQ_ASYNC
#pragma unroll
  for (int j = 0; j < 4; ++j)
    VQ_ASYNC_B128(gsrc + 8 * j, sbuf[0] + 8 * j);
  __builtin_amdgcn_s_wait_asynccnt(0);
#else
  {
    half8 t[4];
#pragma unroll
    for (int j = 0; j < 4; ++j) t[j] = *(const half8*)(gsrc + 8 * j);
#pragma unroll
    for (int j = 0; j < 4; ++j) *(half8*)(sbuf[0] + 8 * j) = t[j];
  }
#endif
  __syncthreads();

  // --- Steady state: unconditional prefetch of tile+1, compute tile --------
  for (int tile = 0; tile + 1 < ntiles; ++tile) {
    const _Float16* gn = gsrc + (size_t)(tile + 1) * 16 * VQ_D;
    _Float16* snext = sbuf[(tile + 1) & 1];

#if VQ_ASYNC
#pragma unroll
    for (int j = 0; j < 4; ++j)
      VQ_ASYNC_B128(gn + 8 * j, snext + 8 * j);
#else
    half8 t[4];
#pragma unroll
    for (int j = 0; j < 4; ++j) t[j] = *(const half8*)(gn + 8 * j);
#endif

    tile_compute(lds_cb + (tile & 1) * 16 * LDS_STRIDE, tile << 4);

#if !VQ_ASYNC
#pragma unroll
    for (int j = 0; j < 4; ++j) *(half8*)(snext + 8 * j) = t[j];
#else
    __builtin_amdgcn_s_wait_asynccnt(0);
#endif
    __syncthreads();
  }

  // --- Epilogue: last tile (no prefetch) -----------------------------------
  tile_compute(lds_cb + ((ntiles - 1) & 1) * 16 * LDS_STRIDE,
               (ntiles - 1) << 4);

  // --- Reduce argmax across the 16 lanes sharing each row (tie -> lower idx)
#pragma unroll
  for (int m = 1; m < 16; m <<= 1) {
#pragma unroll
    for (int v = 0; v < 8; ++v) {
      float ob = __shfl_xor(best[v], m, 32);
      int   oi = __shfl_xor(bidx[v], m, 32);
      if (ob > best[v] || (ob == best[v] && oi < bidx[v])) {
        best[v] = ob; bidx[v] = oi;
      }
    }
  }
  if (n16 == 0) {
#pragma unroll
    for (int v = 0; v < 8; ++v)
      indices[m_base + v + 8 * hi] = bidx[v];
  }
}

// ---------------------------------------------------------------------------
// Kernel 3: gather quantized rows + fused loss = 1.25 * mean((q - x)^2).
// ---------------------------------------------------------------------------
__global__ __launch_bounds__(256) void vq_gather_loss(
    const float* __restrict__ inputs, const float* __restrict__ codebook,
    const int* __restrict__ indices, float* __restrict__ out,
    float* __restrict__ loss, float loss_scale) {
  const long long gid = (long long)blockIdx.x * blockDim.x + threadIdx.x;
  const long long e   = gid * 4;          // element index (grid covers exactly)
  const int b = (int)(e >> 8);
  const int d = (int)(e & 255);
  const int k = indices[b];

  float4 q = *(const float4*)(codebook + (size_t)k * VQ_D + d);
  float4 x = *(const float4*)(inputs + e);
  *(float4*)(out + e) = q;

  float dx = q.x - x.x, dy = q.y - x.y, dz = q.z - x.z, dw = q.w - x.w;
  float s = dx * dx + dy * dy + dz * dz + dw * dw;
#pragma unroll
  for (int m = 1; m < 32; m <<= 1) s += __shfl_xor(s, m, 32);
  if ((threadIdx.x & 31) == 0) atomicAdd(loss, s * loss_scale);
}

// ---------------------------------------------------------------------------
extern "C" void kernel_launch(void* const* d_in, const int* in_sizes, int n_in,
                              void* d_out, int out_size, void* d_ws, size_t ws_size,
                              hipStream_t stream) {
  const float* inputs   = (const float*)d_in[0];
  const float* codebook = (const float*)d_in[1];
  const int Bn = in_sizes[0] / VQ_D;   // 65536
  const int Kn = in_sizes[1] / VQ_D;   // 8192

  // Workspace layout: inp_n f16 | cb_n f16 | indices i32
  _Float16* inp_n = (_Float16*)d_ws;
  _Float16* cb_n  = (_Float16*)((char*)d_ws + (size_t)Bn * VQ_D * sizeof(_Float16));
  int* indices    = (int*)((char*)d_ws + (size_t)Bn * VQ_D * sizeof(_Float16)
                                       + (size_t)Kn * VQ_D * sizeof(_Float16));

  float* out  = (float*)d_out;
  float* loss = out + (size_t)Bn * VQ_D;

  // 1) normalize both operands to f16 (one wave per row, 8 rows per block)
  vq_normalize<<<Bn / 8, 256, 0, stream>>>(inputs, inp_n, Bn);
  vq_normalize<<<(Kn + 7) / 8, 256, 0, stream>>>(codebook, cb_n, Kn);

  // 2) WMMA cosine-similarity sweep + argmax (4 waves/block, 64 rows/block)
  vq_argmax<<<Bn / 64, 128, 0, stream>>>(inp_n, cb_n, indices, Kn);

  // 3) gather + loss
  (void)hipMemsetAsync(loss, 0, sizeof(float), stream);
  const float loss_scale = 1.25f / ((float)Bn * (float)VQ_D);
  const long long n4 = (long long)Bn * VQ_D / 4;
  vq_gather_loss<<<(unsigned)(n4 / 256), 256, 0, stream>>>(
      inputs, codebook, indices, out, loss, loss_scale);
}
